// Model_89610197664142
// MI455X (gfx1250) — matrix operand order repaired and checked by reading, not compiled, not run on hardware
//
#include <hip/hip_runtime.h>
#include <math.h>

typedef _Float16 half_t;
typedef __attribute__((ext_vector_type(16))) _Float16 v16h;
typedef __attribute__((ext_vector_type(8)))  _Float16 v8h;
typedef __attribute__((ext_vector_type(8)))  float    v8f;

#define TPB 256
static inline unsigned gblk(size_t n) { return (unsigned)((n + TPB - 1) / TPB); }

// ---------------- input ordering (setup_inputs dict order) ----------------
enum {
  IN_X = 0,
  IN_D0W, IN_D0B, IN_D1W, IN_D1B, IN_D2W, IN_D2B, IN_D3W, IN_D3B, IN_D4W, IN_D4B,
  IN_U0W, IN_U0B, IN_U1W, IN_U1B, IN_U2W, IN_U2B, IN_U3W, IN_U3B, IN_U4W, IN_U4B,
  IN_PROJW, IN_PROJB,
  IN_ECTXW, IN_ECTXB,
  IN_EOHW, IN_EOHB,
  IN_EMEMW, IN_EMEMB,
  IN_AMPW, IN_AMPB,
  IN_TFLINW, IN_TFLINB,
  IN_TFUP0W, IN_TFUP0B, IN_TFUP1W, IN_TFUP1B,
  IN_TFFINW, IN_TFFINB,
  IN_NMUP0W, IN_NMUP0B, IN_NMUP1W, IN_NMUP1B, IN_NMUP2W, IN_NMUP2B, IN_NMUP3W, IN_NMUP3B,
  IN_NMCOW, IN_NMCOB,
  IN_RFLINW, IN_RFLINB,
  IN_RFUP0W, IN_RFUP0B, IN_RFUP1W, IN_RFUP1B, IN_RFUP2W, IN_RFUP2B, IN_RFUP3W, IN_RFUP3B,
  IN_RFFINW, IN_RFFINB,
  IN_SELW, IN_SELB,
  IN_MOMW, IN_MOMB,
  IN_LNG, IN_LNB,
  IN_ROOMW, IN_ROOMB,
  IN_VMIXW, IN_VMIXB,
  IN_ATOMS, IN_NOISEF, IN_ROOMIRS, IN_REFR
};

// ================= WMMA GEMM =================
// C[M,N] = act(A[M,Kp](f16) * BT[N,Kp]^T(f16) + bias)
// - A row-major, zero-padded to Kp (Kp % 64 == 0); BT pre-transposed [N,Kp]
// - block = 8 waves, 32(M) x 64(N) tile; K tile of 64 staged in LDS via the
//   gfx1250 async global->LDS DMA path (GLOBAL_LOAD_ASYNC_TO_LDS_B128,
//   tracked by ASYNCcnt), then fragments read back with ds_load_b128.
// - staging loads clamp row/col -> no guards in hot loop; guards only at store
template <int ACT>
__global__ void k_gemm_f16(const half_t* __restrict__ A, const half_t* __restrict__ BT,
                           const float* __restrict__ bias, float* __restrict__ C,
                           int M, int N, int Kp) {
  __shared__ half_t As[32 * 64];   // 32 rows x 64 K halves
  __shared__ half_t Bs[64 * 64];   // 64 cols x 64 K halves
  const int tid  = threadIdx.x;
  const int wave = tid >> 5;
  const int lane = tid & 31;
  const int wm = wave >> 2, wn = wave & 3;
  const int blockM = blockIdx.y * 32;
  const int blockN = blockIdx.x * 64;
  // staging: A = 32 rows x 8 segs of 8 halves; B = 64 rows x 4 segs of 16 halves
  const int arow = tid >> 3, aseg = tid & 7;
  const int brow = tid >> 2, bseg = tid & 3;
  const half_t* Ag = A  + (size_t)min(blockM + arow, M - 1) * Kp + aseg * 8;
  const half_t* Bg = BT + (size_t)min(blockN + brow, N - 1) * Kp + bseg * 16;
  // wave-relative LDS byte offsets for the async DMA destination
  const unsigned lAs = (unsigned)(uintptr_t)&As[arow * 64 + aseg * 8];
  const unsigned lBs = (unsigned)(uintptr_t)&Bs[brow * 64 + bseg * 16];
  const int frow = lane & 15;
  const int kA = (lane < 16) ? 0 : 8;    // A frag: K 0-7/16-23 vs 8-15/24-31
  const int kB = (lane < 16) ? 0 : 8;    // B frag: K 0-7/16-23 vs 8-15/24-31
  v8f acc;
#pragma unroll
  for (int r = 0; r < 8; ++r) acc[r] = 0.f;

  for (int k0 = 0; k0 < Kp; k0 += 64) {
    __builtin_prefetch(Ag + k0 + 128, 0, 1);   // global_prefetch_b8, next-next tile
    __builtin_prefetch(Bg + k0 + 128, 0, 1);
    __syncthreads();                           // previous tile fully consumed
    // per-lane async DMA: LDS[lds_off (+16)] = MEM[addr (+16)], 16B each
    asm volatile("global_load_async_to_lds_b128 %0, %1, off"
                 :: "v"(lAs), "v"(Ag + k0) : "memory");
    asm volatile("global_load_async_to_lds_b128 %0, %1, off"
                 :: "v"(lBs), "v"(Bg + k0) : "memory");
    asm volatile("global_load_async_to_lds_b128 %0, %1, off offset:16"
                 :: "v"(lBs), "v"(Bg + k0) : "memory");
    asm volatile("s_wait_asynccnt 0x0" ::: "memory");  // this wave's DMAs landed
    __syncthreads();                           // all waves staged
#pragma unroll
    for (int kk = 0; kk < 64; kk += 32) {
      const half_t* Ar = &As[(wm * 16 + frow) * 64 + kk + kA];
      const half_t* Br = &Bs[(wn * 16 + frow) * 64 + kk + kB];
      v8h a0 = *(const v8h*)(Ar);
      v8h a1 = *(const v8h*)(Ar + 16);
      v8h b0 = *(const v8h*)(Br);
      v8h b1 = *(const v8h*)(Br + 16);
      v16h a = __builtin_shufflevector(a0, a1, 0, 1, 2, 3, 4, 5, 6, 7,
                                       8, 9, 10, 11, 12, 13, 14, 15);
      v16h b = __builtin_shufflevector(b0, b1, 0, 1, 2, 3, 4, 5, 6, 7,
                                       8, 9, 10, 11, 12, 13, 14, 15);
      acc = __builtin_amdgcn_wmma_f32_16x16x32_f16(false, a, false, b, (short)0, acc,
                                                   false, false);
    }
  }
  const int col = blockN + wn * 16 + (lane & 15);
  if (col < N) {
    const float bv = bias ? bias[col] : 0.f;
    const int mb = blockM + wm * 16 + ((lane < 16) ? 0 : 8);
#pragma unroll
    for (int r = 0; r < 8; ++r) {
      int m = mb + r;
      if (m < M) {
        float v = acc[r] + bv;
        if (ACT == 1) v = (v >= 0.f) ? v : 0.2f * v;   // leaky 0.2
        if (ACT == 2) v = fmaxf(v, 0.f);               // relu
        C[(size_t)m * N + col] = v;
      }
    }
  }
}

static void gemm_f16(hipStream_t st, const half_t* A, const half_t* BT, const float* bias,
                     float* C, int M, int N, int Kp, int act) {
  dim3 grid((unsigned)((N + 63) / 64), (unsigned)((M + 31) / 32)), blk(TPB);
  if (act == 1)      k_gemm_f16<1><<<grid, blk, 0, st>>>(A, BT, bias, C, M, N, Kp);
  else if (act == 2) k_gemm_f16<2><<<grid, blk, 0, st>>>(A, BT, bias, C, M, N, Kp);
  else               k_gemm_f16<0><<<grid, blk, 0, st>>>(A, BT, bias, C, M, N, Kp);
}

// ================= small utility kernels =================
__global__ void k_zero_f32(float* p, size_t n) {
  size_t i = (size_t)blockIdx.x * TPB + threadIdx.x; if (i < n) p[i] = 0.f;
}
__global__ void k_copy_f32(const float* a, float* o, size_t n) {
  size_t i = (size_t)blockIdx.x * TPB + threadIdx.x; if (i < n) o[i] = a[i];
}
__global__ void k_add_f32(float* a, const float* b, size_t n) {
  size_t i = (size_t)blockIdx.x * TPB + threadIdx.x; if (i < n) a[i] += b[i];
}
// rows x Kp f16 from rows x K f32, zero-padding the K tail
__global__ void k_pad_f16(const float* a, half_t* o, int K, int Kp, size_t rows) {
  size_t i = (size_t)blockIdx.x * TPB + threadIdx.x;
  size_t tot = rows * (size_t)Kp;
  if (i >= tot) return;
  int k = (int)(i % Kp);
  size_t r = i / Kp;
  o[i] = (k < K) ? (half_t)a[r * K + k] : (half_t)0.0f;
}
// dense [K,N] f32 -> BT [N,Kp] f16
__global__ void k_prep_denseT(const float* w, half_t* o, int K, int N, int Kp) {
  size_t i = (size_t)blockIdx.x * TPB + threadIdx.x;
  size_t tot = (size_t)N * Kp;
  if (i >= tot) return;
  int k = (int)(i % Kp);
  int n = (int)(i / Kp);
  o[i] = (k < K) ? (half_t)w[(size_t)k * N + n] : (half_t)0.0f;
}
// atoms [2046,32768] f32 -> BT [32768,2048] f16 (zero pad K 2046->2048)
__global__ void k_atomsT(const float* a, half_t* o) {
  size_t i = (size_t)blockIdx.x * TPB + threadIdx.x;
  if (i >= (size_t)32768 * 2048) return;
  int k = (int)(i & 2047);
  int n = (int)(i >> 11);
  o[i] = (k < 2046) ? (half_t)a[(size_t)k * 32768 + n] : (half_t)0.0f;
}

// ================= DFT bases (generated transposed, BT layout) =================
__global__ void k_basis_stft_T(half_t* o) {  // BT [2048 bins*2, 2048 samples]
  size_t i = (size_t)blockIdx.x * TPB + threadIdx.x;
  if (i >= (size_t)2048 * 2048) return;
  int s = (int)(i & 2047), c = (int)(i >> 11), q = c >> 1;
  float ang = 6.283185307f * (float)((s * q) & 2047) / 2048.f;
  o[i] = (half_t)((c & 1) ? -__sinf(ang) : __cosf(ang));
}
__global__ void k_basis_fwd_T(half_t* o) {   // BT [1026, 1024]
  size_t i = (size_t)blockIdx.x * TPB + threadIdx.x;
  if (i >= (size_t)1026 * 1024) return;
  int s = (int)(i & 1023), c = (int)(i >> 10), q = c >> 1;
  float ang = 6.283185307f * (float)((s * q) & 1023) / 1024.f;
  o[i] = (half_t)((c & 1) ? -__sinf(ang) : __cosf(ang));
}
__global__ void k_basis_inv_T(half_t* o) {   // BT [1024 t, 1088], irfft incl. 1/N
  size_t i = (size_t)blockIdx.x * TPB + threadIdx.x;
  if (i >= (size_t)1024 * 1088) return;
  int r = (int)(i % 1088), t = (int)(i / 1088);
  if (r >= 1026) { o[i] = (half_t)0.0f; return; }
  int q = r >> 1;
  float w = (q == 0 || q == 512) ? 1.f : 2.f;
  float ang = 6.283185307f * (float)((q * t) & 1023) / 1024.f;
  float v = (r & 1) ? (-w * __sinf(ang)) : (w * __cosf(ang));
  o[i] = (half_t)(v * (1.f / 1024.f));
}

// ================= STFT =================
__global__ void k_frames(const float* x, half_t* fr, int Bn) {  // [B*128, 2048] windowed
  size_t i = (size_t)blockIdx.x * TPB + threadIdx.x;
  size_t tot = (size_t)Bn * 128 * 2048;
  if (i >= tot) return;
  int j = (int)(i % 2048), f = (int)((i / 2048) % 128), b = (int)(i / (2048 * 128));
  int pos = f * 256 + j;
  float v = (pos < 32768) ? x[(size_t)b * 32768 + pos] : 0.f;
  float h = 0.5f - 0.5f * __cosf(6.283185307f * (float)j / 2048.f);
  fr[i] = (half_t)(v * h);
}
__global__ void k_stft_mag(const float* S, float* h, int Bn) {  // -> NCL [B,1024,128]
  size_t i = (size_t)blockIdx.x * TPB + threadIdx.x;
  size_t tot = (size_t)Bn * 1024 * 128;
  if (i >= tot) return;
  int t = (int)(i % 128), c = (int)((i / 128) % 1024), b = (int)(i / (128 * 1024));
  const float* row = S + ((size_t)(b * 128 + t)) * 2048;
  float re = row[2 * c], im = row[2 * c + 1];
  h[i] = sqrtf(re * re + im * im);
}

// ================= conv plumbing =================
// weights [Cout,Cin,kW] f32 -> BT [Cout, Cin*kW] f16 (optionally kernel-flipped)
__global__ void k_prep_convw(const float* w, half_t* o, int Cout, int Cin, int kW, int flip) {
  size_t i = (size_t)blockIdx.x * TPB + threadIdx.x;
  size_t tot = (size_t)Cout * Cin * kW;
  if (i >= tot) return;
  int co = (int)(i / ((size_t)Cin * kW));
  int r  = (int)(i % ((size_t)Cin * kW));
  int ci = r / kW, k = r % kW;
  int ks = flip ? (kW - 1 - k) : k;
  o[i] = (half_t)w[((size_t)co * Cin + ci) * kW + ks];
}
__global__ void k_im2col_conv(const float* h, half_t* cols, int nb, int Cin, int Lin,
                              int Lout, int kW, int stride, int pad) {
  size_t i = (size_t)blockIdx.x * TPB + threadIdx.x;
  size_t tot = (size_t)nb * Lout * Cin * kW;
  if (i >= tot) return;
  int kk = (int)(i % kW);
  int ci = (int)((i / kW) % Cin);
  size_t rt = i / ((size_t)Cin * kW);
  int t = (int)(rt % Lout);
  int n = (int)(rt / Lout);
  int pos = t * stride + kk - pad;
  float v = (pos >= 0 && pos < Lin) ? h[((size_t)n * Cin + ci) * Lin + pos] : 0.f;
  cols[i] = (half_t)v;
}
__global__ void k_im2col_convT(const float* h, half_t* cols, int nb, int Cin, int Lin) {
  // ConvTranspose1d k=4 s=2 p=1: gather from dilated input, weights pre-flipped
  size_t i = (size_t)blockIdx.x * TPB + threadIdx.x;
  int Lout = 2 * Lin;
  size_t tot = (size_t)nb * Lout * Cin * 4;
  if (i >= tot) return;
  int kk = (int)(i & 3);
  int ci = (int)((i >> 2) % Cin);
  size_t rt = i / ((size_t)Cin * 4);
  int t = (int)(rt % Lout);
  int n = (int)(rt / Lout);
  int pos = t + kk - 2;
  float v = 0.f;
  if (pos >= 0 && !(pos & 1) && (pos >> 1) < Lin)
    v = h[((size_t)n * Cin + ci) * Lin + (pos >> 1)];
  cols[i] = (half_t)v;
}
__global__ void k_nlc_to_ncl(const float* y, float* h, int nb, int C, int L) {
  size_t i = (size_t)blockIdx.x * TPB + threadIdx.x;
  size_t tot = (size_t)nb * C * L;
  if (i >= tot) return;
  int t = (int)(i % L), c = (int)((i / L) % C), n = (int)(i / ((size_t)L * C));
  h[i] = y[((size_t)n * L + t) * C + c];
}

// ================= refractory FIR + top-k + events =================
__global__ void k_refractory(const float* enc, const float* refr, float* o, int tot) {
  int i = blockIdx.x * TPB + threadIdx.x;
  if (i >= tot) return;
  int t = i & 127;
  float acc = 0.f;
#pragma unroll
  for (int s = 0; s < 8; ++s)
    if (s <= t) acc += enc[i - s] * refr[s];
  o[i] = acc;
}
__global__ void k_topk(float* flat, float* vals, int* idxs, int Nflat) {
  int b = blockIdx.x, tid = threadIdx.x;
  flat += (size_t)b * Nflat;
  __shared__ float sv[TPB];
  __shared__ int   si[TPB];
  for (int it = 0; it < 64; ++it) {
    float bv = -3.4e38f; int bi = 0;
    for (int i = tid; i < Nflat; i += TPB) {
      float v = flat[i];
      if (v > bv || (v == bv && i < bi)) { bv = v; bi = i; }
    }
    sv[tid] = bv; si[tid] = bi;
    __syncthreads();
    for (int s = TPB / 2; s > 0; s >>= 1) {
      if (tid < s) {
        if (sv[tid + s] > sv[tid] || (sv[tid + s] == sv[tid] && si[tid + s] < si[tid])) {
          sv[tid] = sv[tid + s]; si[tid] = si[tid + s];
        }
      }
      __syncthreads();
    }
    if (tid == 0) {
      vals[b * 64 + it] = sv[0];
      idxs[b * 64 + it] = si[0];
      flat[si[0]] = -3.4e38f;
    }
    __syncthreads();
  }
}
__global__ void k_events(const float* vals, const int* idxs, float* encoded, float* ctxt,
                         int* chA, int* tA, int Bn) {
  int i = blockIdx.x * TPB + threadIdx.x;
  if (i >= Bn * 64) return;
  int b = i >> 6;
  int id = idxs[i];
  int ch = id / 128, tt = id & 127;
  chA[i] = ch; tA[i] = tt;
  float v = vals[i];
  float rv = v > 0.f ? v : 0.f;
  encoded[((size_t)b * 4096 + ch) * 128 + tt] = rv;
  atomicAdd(&ctxt[b * 4096 + ch], rv);
}

// ================= small dense layers =================
__global__ void k_linear_f32(const float* A, const float* W, const float* bias, float* C,
                             int M, int K, int N, int act) {
  int i = blockIdx.x * TPB + threadIdx.x;
  if (i >= M * N) return;
  int m = i / N, n = i % N;
  float acc = bias ? bias[n] : 0.f;
  for (int k = 0; k < K; ++k) acc += A[(size_t)m * K + k] * W[(size_t)k * N + n];
  if (act == 1) acc = acc >= 0.f ? acc : 0.2f * acc;
  else if (act == 2) acc = fmaxf(acc, 0.f);
  else if (act == 3) acc = fabsf(acc);
  C[i] = acc;
}
__global__ void k_oh_embed(const float* ce, const int* chA, const float* ohW, const float* ohB,
                           float* emb, int NE) {
  int i = blockIdx.x * TPB + threadIdx.x;
  if (i >= NE * 256) return;
  int c = i & 255, n = i >> 8, b = n >> 6;
  emb[i] = ce[b * 256 + c] + ohW[(size_t)chA[n] * 256 + c] + ohB[c];
}
__global__ void k_mom(const float* pre, float* mom, int NE) {
  int n = blockIdx.x * TPB + threadIdx.x;
  if (n >= NE) return;
  float run = 0.f;
  for (int j = 0; j < 64; ++j) {
    float s = 1.f / (1.f + __expf(-pre[n * 64 + j]));
    float m = 0.02f + s * 0.931f;            // BASE_RES + sig * (1-BASE)*0.95
    run += __logf(1e-12f + m);
    mom[n * 64 + j] = __expf(run);
  }
}

// ================= noise model =================
__global__ void k_filt(const float* y, float* filt, int NE) {  // NCL [NE,17,256]
  size_t i = (size_t)blockIdx.x * TPB + threadIdx.x;
  size_t tot = (size_t)NE * 17 * 256;
  if (i >= tot) return;
  int c = (int)((i / 256) % 17);
  float s = 1.f / (1.f + __expf(-y[i]));
  filt[i] = (c >= 1) ? s * s : 0.f;
}
__global__ void k_noise_spec(const float* nf, float* sp) {  // rfft32 of [256,32]
  int i = blockIdx.x * TPB + threadIdx.x;
  if (i >= 256 * 17) return;
  int q = i % 17, f = i / 17;
  float re = 0.f, im = 0.f;
  for (int t = 0; t < 32; ++t) {
    float v = nf[f * 32 + t];
    float ang = 6.283185307f * (float)((q * t) & 31) / 32.f;
    float s, c; __sincosf(ang, &s, &c);
    re += v * c; im -= v * s;
  }
  sp[i * 2] = re; sp[i * 2 + 1] = im;
}
__global__ void k_impulse(const float* filt, const float* nspec, float* imp, int NE) {
  size_t i = (size_t)blockIdx.x * TPB + threadIdx.x;
  if (i >= (size_t)NE * 4096) return;
  int n = (int)(i >> 12), s = (int)(i & 4095);
  float acc = 0.f;
  int f0 = s >> 4;
  for (int f = f0 - 1; f <= f0; ++f) {
    if (f < 0 || f >= 256) continue;
    int t = s - (f << 4);
    if (t >= 32) continue;
    float sum = 0.f;
    for (int k = 1; k <= 16; ++k) {   // DC masked
      float fr = filt[((size_t)n * 17 + k) * 256 + f];
      float re = nspec[(f * 17 + k) * 2 + 0] * fr;
      float im = nspec[(f * 17 + k) * 2 + 1] * fr;
      float w = (k == 16) ? 1.f : 2.f;
      float ang = 6.283185307f * (float)((k * t) & 31) / 32.f;
      float sn, cs; __sincosf(ang, &sn, &cs);
      sum += w * (re * cs - im * sn);
    }
    acc += sum * (1.f / 32.f);
  }
  imp[i] = acc;
}

// ================= resonance =================
__global__ void k_rfilt(const float* g, float* rf, int NE) {  // g NCL [NE,32,64]
  size_t i = (size_t)blockIdx.x * TPB + threadIdx.x;
  size_t tot = (size_t)NE * 64 * 513;
  if (i >= tot) return;
  int q = (int)(i % 513), j = (int)((i / 513) % 64), n = (int)(i / (513 * 64));
  float pos = (q + 0.5f) * (32.f / 513.f) - 0.5f;
  pos = fminf(fmaxf(pos, 0.f), 31.f);
  int lo = (int)pos;
  int hi = (lo + 1 > 31) ? 31 : lo + 1;
  float w = pos - (float)lo;
  float v = g[((size_t)n * 32 + lo) * 64 + j] * (1.f - w) +
            g[((size_t)n * 32 + hi) * 64 + j] * w;
  rf[i] = 1.f / (1.f + __expf(-v));
}
__global__ void k_window(const float* res, const float* mom, half_t* winh, int NE) {
  int nf = blockIdx.x, tid = threadIdx.x;      // NE*64 blocks
  int n = nf >> 6, f = nf & 63;
  float v[4], ss = 0.f;
#pragma unroll
  for (int j = 0; j < 4; ++j) {
    int sl = tid + j * 256;
    int gp = f * 512 + sl;
    float x = (gp < 32768) ? res[(size_t)n * 32768 + gp] : 0.f;
    float h = 0.5f - 0.5f * __cosf(6.283185307f * (float)sl / 1024.f);
    v[j] = x * h;
    ss += v[j] * v[j];
  }
  __shared__ float red[TPB];
  red[tid] = ss;
  __syncthreads();
  for (int s = TPB / 2; s > 0; s >>= 1) {
    if (tid < s) red[tid] += red[tid + s];
    __syncthreads();
  }
  float scale = mom[n * 64 + f] / (sqrtf(red[0]) + 1e-8f);
#pragma unroll
  for (int j = 0; j < 4; ++j)
    winh[(size_t)nf * 1024 + tid + j * 256] = (half_t)(v[j] * scale);
}
__global__ void k_specmul(float* spec, const float* rf, int rows) {
  size_t i = (size_t)blockIdx.x * TPB + threadIdx.x;
  if (i >= (size_t)rows * 1026) return;
  int q = (int)((i % 1026) >> 1);
  size_t r = i / 1026;
  spec[i] *= rf[r * 513 + q];
}
__global__ void k_ola512(const float* fr, float* sig, int NE) {
  size_t i = (size_t)blockIdx.x * TPB + threadIdx.x;
  if (i >= (size_t)NE * 32768) return;
  int n = (int)(i >> 15), s = (int)(i & 32767);
  float acc = 0.f;
  int f0 = s >> 9;
  for (int f = f0 - 1; f <= f0; ++f) {
    if (f < 0 || f >= 64) continue;
    int tl = s - (f << 9);
    if (tl < 1024) acc += fr[((size_t)n * 64 + f) * 1024 + tl];
  }
  sig[i] = acc;
}

// ================= mix / placement / reverb =================
__global__ void k_mixconv(const float* imp, const float* sig, const float* amp, float* mixed) {
  int n = blockIdx.y;
  int sb = blockIdx.x * 256, tid = threadIdx.x, s = sb + tid;
  __shared__ float li[256];
  __shared__ float lr[512];
  float acc = 0.f;
  for (int ut = 0; ut < 4096; ut += 256) {
    li[tid] = imp[(size_t)n * 4096 + ut + tid];
    int base = sb - ut - 255;
    int i1 = base + tid;        lr[tid]       = (i1 >= 0 && i1 < 32768) ? sig[(size_t)n * 32768 + i1] : 0.f;
    int i2 = base + 256 + tid;  lr[256 + tid] = (i2 >= 0 && i2 < 32768) ? sig[(size_t)n * 32768 + i2] : 0.f;
    __syncthreads();
#pragma unroll 8
    for (int uu = 0; uu < 256; ++uu) acc += li[uu] * lr[tid - uu + 255];
    __syncthreads();
  }
  float pad = (s < 4096) ? imp[(size_t)n * 4096 + s] : 0.f;
  mixed[(size_t)n * 32768 + s] = (acc + pad) * amp[n];
}
__global__ void k_dry(const float* mixed, const float* vals, const int* tA, float* dry, int Bn) {
  int i = blockIdx.x * TPB + threadIdx.x;
  if (i >= Bn * 32768) return;
  int s = i & 32767, b = i >> 15;
  float a = 0.f;
  for (int e = 0; e < 64; ++e) {
    int n = b * 64 + e;
    int sh = s - (tA[n] << 8);
    if (sh >= 0) a += vals[n] * mixed[(size_t)n * 32768 + sh];
  }
  dry[i] = a;
}
__global__ void k_reverb_setup(const float* dense, const float* lng, const float* lnb,
                               const float* roomw, const float* roomb,
                               const float* vmixw, const float* vmixb,
                               float* rsel, float* wsel, int Bn) {
  int b = blockIdx.x * blockDim.x + threadIdx.x;
  if (b >= Bn) return;
  float d[16], mu = 0.f;
  for (int i = 0; i < 16; ++i) { d[i] = dense[b * 16 + i]; mu += d[i]; }
  mu *= (1.f / 16.f);
  float var = 0.f;
  for (int i = 0; i < 16; ++i) { float t = d[i] - mu; var += t * t; }
  var *= (1.f / 16.f);
  float inv = rsqrtf(var + 1e-5f);
  for (int i = 0; i < 16; ++i) d[i] = (d[i] - mu) * inv * lng[i] + lnb[i];
  float r[8], mx = -3.4e38f;
  for (int n = 0; n < 8; ++n) {
    float a = roomb[n];
    for (int i = 0; i < 16; ++i) a += d[i] * roomw[i * 8 + n];
    r[n] = a; mx = fmaxf(mx, a);
  }
  float ssum = 0.f;
  for (int n = 0; n < 8; ++n) { r[n] = __expf(r[n] - mx); ssum += r[n]; }
  for (int n = 0; n < 8; ++n) rsel[b * 8 + n] = r[n] / ssum;
  float w2[2]; mx = -3.4e38f;
  for (int n = 0; n < 2; ++n) {
    float a = vmixb[n];
    for (int i = 0; i < 16; ++i) a += d[i] * vmixw[i * 2 + n];
    w2[n] = a; mx = fmaxf(mx, a);
  }
  ssum = 0.f;
  for (int n = 0; n < 2; ++n) { w2[n] = __expf(w2[n] - mx); ssum += w2[n]; }
  for (int n = 0; n < 2; ++n) wsel[b * 2 + n] = w2[n] / ssum;
}
__global__ void k_mk_ir(const float* rsel, const float* irs, float* ir, int Bn) {
  int i = blockIdx.x * TPB + threadIdx.x;
  if (i >= Bn * 32768) return;
  int s = i & 32767, b = i >> 15;
  float a = 0.f;
  for (int r = 0; r < 8; ++r) a += rsel[b * 8 + r] * irs[(size_t)r * 32768 + s];
  ir[i] = a;
}
__global__ void k_wet(const float* dry, const float* ir, const float* wsel, float* out) {
  int b = blockIdx.y;
  int sb = blockIdx.x * 256, tid = threadIdx.x, s = sb + tid;
  const float* drb = dry + (size_t)b * 32768;
  const float* irb = ir + (size_t)b * 32768;
  __shared__ float ld[256];
  __shared__ float li[512];
  float acc = 0.f;
  for (int ut = 0; ut < 32768; ut += 256) {
    ld[tid] = drb[ut + tid];
    int base = sb - ut - 255;
    int i1 = base + tid;        li[tid]       = (i1 >= 0 && i1 < 32768) ? irb[i1] : 0.f;
    int i2 = base + 256 + tid;  li[256 + tid] = (i2 >= 0 && i2 < 32768) ? irb[i2] : 0.f;
    __syncthreads();
#pragma unroll 8
    for (int uu = 0; uu < 256; ++uu) acc += ld[uu] * li[tid - uu + 255];
    __syncthreads();
  }
  out[(size_t)b * 32768 + s] = wsel[b * 2] * drb[s] + wsel[b * 2 + 1] * acc;
}

// ======================= orchestration =======================
extern "C" void kernel_launch(void* const* d_in, const int* in_sizes, int n_in,
                              void* d_out, int out_size, void* d_ws, size_t ws_size,
                              hipStream_t stream) {
  (void)n_in; (void)out_size; (void)ws_size;
  const int B = in_sizes[0] / 32768;
  const int NE = B * 64;

  auto P = [&](int i) { return (const float*)d_in[i]; };
  float* out = (float*)d_out;
  float* enc_out = out + (size_t)B * 32768;
  float* imp_out = enc_out + (size_t)B * 4096 * 128;

  // ---- workspace bump allocator (deterministic layout, rebuilt each call) ----
  char* wsp = (char*)d_ws;
  size_t wof = 0;
  auto alloc = [&](size_t bytes) -> void* {
    size_t o = (wof + 255) & ~(size_t)255;
    wof = o + bytes;
    return wsp + o;
  };
  half_t* basisS  = (half_t*)alloc((size_t)2048 * 2048 * 2);
  half_t* basisF  = (half_t*)alloc((size_t)1026 * 1024 * 2);
  half_t* basisI  = (half_t*)alloc((size_t)1024 * 1088 * 2);
  half_t* frames  = (half_t*)alloc((size_t)B * 128 * 2048 * 2);
  half_t* cols    = (half_t*)alloc((size_t)16777216 * 2);
  half_t* wf16    = (half_t*)alloc((size_t)4194304 * 2);
  half_t* atoms16 = (half_t*)alloc((size_t)32768 * 2048 * 2);
  half_t* spec16  = (half_t*)alloc((size_t)NE * 64 * 1088 * 2);
  half_t* embf16  = (half_t*)alloc((size_t)NE * 256 * 2);
  half_t* self16  = (half_t*)alloc((size_t)NE * 2048 * 2);

  float* stftS  = (float*)alloc((size_t)B * 128 * 2048 * 4);
  float* p1     = (float*)alloc((size_t)4194304 * 4);   // GEMM NLC outputs
  float* p2     = (float*)alloc((size_t)4194304 * 4);   // NCL feature maps
  float* ctxb[4];
  const int ctxL[4] = {64, 32, 16, 8};
  for (int j = 0; j < 4; ++j) ctxb[j] = (float*)alloc((size_t)B * 1024 * ctxL[j] * 4);
  float* flat   = (float*)alloc((size_t)B * 4096 * 128 * 4);
  float* vals   = (float*)alloc((size_t)NE * 4);
  int*   idxs   = (int*)alloc((size_t)NE * 4);
  int*   chA    = (int*)alloc((size_t)NE * 4);
  int*   tA     = (int*)alloc((size_t)NE * 4);
  float* ctxt   = (float*)alloc((size_t)B * 4096 * 4);
  float* densev = (float*)alloc((size_t)B * 16 * 4);
  float* ce     = (float*)alloc((size_t)B * 256 * 4);
  float* emb    = (float*)alloc((size_t)NE * 256 * 4);
  float* amp    = (float*)alloc((size_t)NE * 4);
  float* mompre = (float*)alloc((size_t)NE * 64 * 4);
  float* momv   = (float*)alloc((size_t)NE * 64 * 4);
  float* filtb  = (float*)alloc((size_t)NE * 17 * 256 * 4);
  float* nspec  = (float*)alloc((size_t)256 * 17 * 2 * 4);
  float* selb   = (float*)alloc((size_t)NE * 2046 * 4);
  float* resb   = (float*)alloc((size_t)NE * 32768 * 4);
  float* rfilt  = (float*)alloc((size_t)NE * 64 * 513 * 4);
  float* big1   = (float*)alloc((size_t)NE * 64 * 1026 * 4);
  float* big2   = (float*)alloc((size_t)NE * 64 * 1024 * 4);
  float* ressig = (float*)alloc((size_t)NE * 32768 * 4);
  float* mixed  = (float*)alloc((size_t)NE * 32768 * 4);
  float* dry    = (float*)alloc((size_t)B * 32768 * 4);
  float* irb    = (float*)alloc((size_t)B * 32768 * 4);
  float* rsel   = (float*)alloc((size_t)B * 8 * 4);
  float* wsel   = (float*)alloc((size_t)B * 2 * 4);

  // ---- conv helpers: read NCL p2, result NCL back in p2 ----
  auto do_conv = [&](int nb, int Cin, int Cout, int Lin, int kW, int stride, int pad,
                     const float* w, const float* bias, int act) {
    int Lout = (Lin + 2 * pad - kW) / stride + 1;
    k_prep_convw<<<gblk((size_t)Cout * Cin * kW), TPB, 0, stream>>>(w, wf16, Cout, Cin, kW, 0);
    k_im2col_conv<<<gblk((size_t)nb * Lout * Cin * kW), TPB, 0, stream>>>(
        p2, cols, nb, Cin, Lin, Lout, kW, stride, pad);
    gemm_f16(stream, cols, wf16, bias, p1, nb * Lout, Cout, Cin * kW, act);
    k_nlc_to_ncl<<<gblk((size_t)nb * Cout * Lout), TPB, 0, stream>>>(p1, p2, nb, Cout, Lout);
  };
  auto do_convT = [&](int nb, int Cin, int Cout, int Lin,
                      const float* w, const float* bias, int act) {
    int Lout = 2 * Lin;
    k_prep_convw<<<gblk((size_t)Cout * Cin * 4), TPB, 0, stream>>>(w, wf16, Cout, Cin, 4, 1);
    k_im2col_convT<<<gblk((size_t)nb * Lout * Cin * 4), TPB, 0, stream>>>(p2, cols, nb, Cin, Lin);
    gemm_f16(stream, cols, wf16, bias, p1, nb * Lout, Cout, Cin * 4, act);
    k_nlc_to_ncl<<<gblk((size_t)nb * Cout * Lout), TPB, 0, stream>>>(p1, p2, nb, Cout, Lout);
  };

  // ================= 1) STFT as DFT GEMM =================
  k_basis_stft_T<<<gblk((size_t)2048 * 2048), TPB, 0, stream>>>(basisS);
  k_basis_fwd_T<<<gblk((size_t)1026 * 1024), TPB, 0, stream>>>(basisF);
  k_basis_inv_T<<<gblk((size_t)1024 * 1088), TPB, 0, stream>>>(basisI);
  k_frames<<<gblk((size_t)B * 128 * 2048), TPB, 0, stream>>>(P(IN_X), frames, B);
  gemm_f16(stream, frames, basisS, nullptr, stftS, B * 128, 2048, 2048, 0);
  k_stft_mag<<<gblk((size_t)B * 1024 * 128), TPB, 0, stream>>>(stftS, p2, B);

  // ================= 2) UNet =================
  const int DWI[5] = {IN_D0W, IN_D1W, IN_D2W, IN_D3W, IN_D4W};
  const int DBI[5] = {IN_D0B, IN_D1B, IN_D2B, IN_D3B, IN_D4B};
  const int UWI[5] = {IN_U0W, IN_U1W, IN_U2W, IN_U3W, IN_U4W};
  const int UBI[5] = {IN_U0B, IN_U1B, IN_U2B, IN_U3B, IN_U4B};
  int L = 128;
  for (int i = 0; i < 5; ++i) {
    do_conv(B, 1024, 1024, L, 3, 2, 1, P(DWI[i]), P(DBI[i]), 1);
    L /= 2;
    if (i < 4)
      k_copy_f32<<<gblk((size_t)B * 1024 * L), TPB, 0, stream>>>(p2, ctxb[i], (size_t)B * 1024 * L);
  }
  for (int i = 0; i < 5; ++i) {
    do_convT(B, 1024, 1024, L, P(UWI[i]), P(UBI[i]), 1);
    L *= 2;
    if (i < 4)
      k_add_f32<<<gblk((size_t)B * 1024 * L), TPB, 0, stream>>>(p2, ctxb[3 - i], (size_t)B * 1024 * L);
  }
  // proj (k=1) + refractory FIR
  do_conv(B, 1024, 4096, 128, 1, 1, 0, P(IN_PROJW), P(IN_PROJB), 0);
  k_refractory<<<gblk((size_t)B * 4096 * 128), TPB, 0, stream>>>(p2, P(IN_REFR), flat, B * 4096 * 128);

  // ================= 3) top-k events + embeddings =================
  k_topk<<<dim3(B), dim3(TPB), 0, stream>>>(flat, vals, idxs, 4096 * 128);
  k_zero_f32<<<gblk((size_t)B * 4096 * 128), TPB, 0, stream>>>(enc_out, (size_t)B * 4096 * 128);
  k_zero_f32<<<gblk((size_t)B * 4096), TPB, 0, stream>>>(ctxt, (size_t)B * 4096);
  k_events<<<gblk((size_t)NE), TPB, 0, stream>>>(vals, idxs, enc_out, ctxt, chA, tA, B);
  k_linear_f32<<<gblk((size_t)B * 16), TPB, 0, stream>>>(ctxt, P(IN_EMEMW), P(IN_EMEMB), densev, B, 4096, 16, 0);
  k_linear_f32<<<gblk((size_t)B * 256), TPB, 0, stream>>>(ctxt, P(IN_ECTXW), P(IN_ECTXB), ce, B, 4096, 256, 0);
  k_oh_embed<<<gblk((size_t)NE * 256), TPB, 0, stream>>>(ce, chA, P(IN_EOHW), P(IN_EOHB), emb, NE);
  k_pad_f16<<<gblk((size_t)NE * 256), TPB, 0, stream>>>(emb, embf16, 256, 256, (size_t)NE);
  k_linear_f32<<<gblk((size_t)NE), TPB, 0, stream>>>(emb, P(IN_AMPW), P(IN_AMPB), amp, NE, 256, 1, 3);
  k_linear_f32<<<gblk((size_t)NE * 64), TPB, 0, stream>>>(emb, P(IN_MOMW), P(IN_MOMB), mompre, NE, 256, 64, 0);
  k_mom<<<gblk((size_t)NE), TPB, 0, stream>>>(mompre, momv, NE);

  // ================= 4) impulse path (tf + noise model) =================
  k_prep_denseT<<<gblk((size_t)512 * 256), TPB, 0, stream>>>(P(IN_TFLINW), wf16, 256, 512, 256);
  gemm_f16(stream, embf16, wf16, P(IN_TFLINB), p2, NE, 512, 256, 0);   // NCL [NE,128,4]
  do_convT(NE, 128, 128, 4, P(IN_TFUP0W), P(IN_TFUP0B), 1);
  do_convT(NE, 128, 128, 8, P(IN_TFUP1W), P(IN_TFUP1B), 1);
  do_conv(NE, 128, 128, 16, 3, 1, 1, P(IN_TFFINW), P(IN_TFFINB), 0);
  const int NMW[4] = {IN_NMUP0W, IN_NMUP1W, IN_NMUP2W, IN_NMUP3W};
  const int NMB[4] = {IN_NMUP0B, IN_NMUP1B, IN_NMUP2B, IN_NMUP3B};
  int Ln = 16;
  for (int i = 0; i < 4; ++i) { do_convT(NE, 128, 128, Ln, P(NMW[i]), P(NMB[i]), 1); Ln *= 2; }
  do_conv(NE, 128, 17, 256, 3, 1, 1, P(IN_NMCOW), P(IN_NMCOB), 0);
  k_filt<<<gblk((size_t)NE * 17 * 256), TPB, 0, stream>>>(p2, filtb, NE);
  k_noise_spec<<<gblk((size_t)256 * 17), TPB, 0, stream>>>(P(IN_NOISEF), nspec);
  k_impulse<<<gblk((size_t)NE * 4096), TPB, 0, stream>>>(filtb, nspec, imp_out, NE);

  // ================= 5) resonance path =================
  k_prep_denseT<<<gblk((size_t)256 * 256), TPB, 0, stream>>>(P(IN_RFLINW), wf16, 256, 256, 256);
  gemm_f16(stream, embf16, wf16, P(IN_RFLINB), p2, NE, 256, 256, 0);   // NCL [NE,64,4]
  const int RFW[4] = {IN_RFUP0W, IN_RFUP1W, IN_RFUP2W, IN_RFUP3W};
  const int RFB[4] = {IN_RFUP0B, IN_RFUP1B, IN_RFUP2B, IN_RFUP3B};
  int Lr = 4;
  for (int i = 0; i < 4; ++i) { do_convT(NE, 64, 64, Lr, P(RFW[i]), P(RFB[i]), 1); Lr *= 2; }
  do_conv(NE, 64, 32, 64, 3, 1, 1, P(IN_RFFINW), P(IN_RFFINB), 0);
  k_rfilt<<<gblk((size_t)NE * 64 * 513), TPB, 0, stream>>>(p2, rfilt, NE);

  // sel @ atoms : the dominant GEMM (f16 WMMA, fp32 accumulate)
  k_prep_denseT<<<gblk((size_t)2046 * 256), TPB, 0, stream>>>(P(IN_SELW), wf16, 256, 2046, 256);
  gemm_f16(stream, embf16, wf16, P(IN_SELB), selb, NE, 2046, 256, 2);
  k_pad_f16<<<gblk((size_t)NE * 2048), TPB, 0, stream>>>(selb, self16, 2046, 2048, (size_t)NE);
  k_atomsT<<<gblk((size_t)32768 * 2048), TPB, 0, stream>>>(P(IN_ATOMS), atoms16);
  gemm_f16(stream, self16, atoms16, nullptr, resb, NE, 32768, 2048, 0);

  // window -> rfft (GEMM) -> *rfilt -> irfft (GEMM) -> overlap-add
  k_window<<<dim3((unsigned)(NE * 64)), dim3(TPB), 0, stream>>>(resb, momv, (half_t*)cols, NE);
  gemm_f16(stream, (half_t*)cols, basisF, nullptr, big1, NE * 64, 1026, 1024, 0);
  k_specmul<<<gblk((size_t)NE * 64 * 1026), TPB, 0, stream>>>(big1, rfilt, NE * 64);
  k_pad_f16<<<gblk((size_t)NE * 64 * 1088), TPB, 0, stream>>>(big1, spec16, 1026, 1088, (size_t)NE * 64);
  gemm_f16(stream, spec16, basisI, nullptr, big2, NE * 64, 1024, 1088, 0);
  k_ola512<<<gblk((size_t)NE * 32768), TPB, 0, stream>>>(big2, ressig, NE);

  // ================= 6) mix, event placement, reverb =================
  k_mixconv<<<dim3(128, (unsigned)NE), dim3(TPB), 0, stream>>>(imp_out, ressig, amp, mixed);
  k_dry<<<gblk((size_t)B * 32768), TPB, 0, stream>>>(mixed, vals, tA, dry, B);
  k_reverb_setup<<<dim3(1), dim3(32), 0, stream>>>(densev, P(IN_LNG), P(IN_LNB),
                                                   P(IN_ROOMW), P(IN_ROOMB),
                                                   P(IN_VMIXW), P(IN_VMIXB), rsel, wsel, B);
  k_mk_ir<<<gblk((size_t)B * 32768), TPB, 0, stream>>>(rsel, P(IN_ROOMIRS), irb, B);
  k_wet<<<dim3(128, (unsigned)B), dim3(TPB), 0, stream>>>(dry, irb, wsel, out);
}
